// SparseKernelMultiheadAttention_12747462935035
// MI455X (gfx1250) — matrix operand order repaired and checked by reading, not compiled, not run on hardware
//
#include <hip/hip_runtime.h>
#include <hip/hip_bf16.h>

typedef __attribute__((ext_vector_type(16))) _Float16 v16h;
typedef __attribute__((ext_vector_type(8)))  float    v8f;
typedef __attribute__((ext_vector_type(8)))  unsigned int v8u;

#define NTOK  1024
#define EMB   1024
#define HEADS 16
#define DH    64
#define SS    128

// ---------------------------------------------------------------------------
// CDNA5 async-copy helpers (global -> LDS direct path, ASYNCcnt-tracked)
// ---------------------------------------------------------------------------
__device__ __forceinline__ void async_copy16(unsigned lds_addr, const void* gptr) {
  asm volatile("global_load_async_to_lds_b128 %0, %1, off"
               :: "v"(lds_addr), "v"((unsigned long long)(size_t)gptr)
               : "memory");
}
__device__ __forceinline__ void wait_async0() {
  asm volatile("s_wait_asynccnt 0x0" ::: "memory");
}

// ---------------------------------------------------------------------------
// f32 -> f16 elementwise conversion
// ---------------------------------------------------------------------------
__global__ void cvt_f32_to_f16(const float* __restrict__ src,
                               _Float16* __restrict__ dst, int n) {
  int i = blockIdx.x * blockDim.x + threadIdx.x;
  if (i < n) dst[i] = (_Float16)src[i];
}

// ---------------------------------------------------------------------------
// WMMA GEMM: C[M x Nn] = A[M x K] (f16 row-major) @ W[Nn x K]^T + bias
// 128x128 block tile, BK=32, 8 waves (4x2), each wave 32x64 out =
// 2x4 v_wmma_f32_16x16x32_f16. Double-buffered LDS fed by
// global_load_async_to_lds_b128 (no VGPR round-trip).
// ---------------------------------------------------------------------------
#define BM 128
#define BN 128
#define BK 32

__global__ __launch_bounds__(256) void gemm_wmma_f16(
    const _Float16* __restrict__ A, const _Float16* __restrict__ W,
    const float* __restrict__ bias, float* __restrict__ C,
    int M, int Nn, int K)
{
  __shared__ __align__(16) _Float16 As[2][BM][BK];   // 2 x 8 KB
  __shared__ __align__(16) _Float16 Bs[2][BN][BK];   // 2 x 8 KB

  const int tid  = threadIdx.x;
  const int lane = tid & 31;
  const int wid  = tid >> 5;
  const int wm   = wid >> 1;          // 0..3 -> 32 rows each
  const int wn   = wid & 1;           // 0..1 -> 64 cols each
  const int m0   = blockIdx.y * BM;
  const int n0   = blockIdx.x * BN;

  const int r16 = lane & 15;          // row/col within 16-tile
  const int hi  = lane >> 4;          // lane-group select

  // per-thread cooperative-copy coordinates: 2 x 16B chunks per tile
  const int idx0  = tid;              // 0..255
  const int idx1  = tid + 256;        // 256..511
  const int rowL0 = idx0 >> 2, cL0 = (idx0 & 3) * 8;   // 4 chunks per 32-half row
  const int rowL1 = idx1 >> 2, cL1 = (idx1 & 3) * 8;

  v8f acc[2][4] = {};                 // zero-initialized accumulators

#define ISSUE_TILE(buf, kb)                                                    \
  do {                                                                         \
    async_copy16((unsigned)(size_t)&As[buf][rowL0][cL0],                       \
                 A + (size_t)(m0 + rowL0) * K + (kb) + cL0);                   \
    async_copy16((unsigned)(size_t)&As[buf][rowL1][cL1],                       \
                 A + (size_t)(m0 + rowL1) * K + (kb) + cL1);                   \
    async_copy16((unsigned)(size_t)&Bs[buf][rowL0][cL0],                       \
                 W + (size_t)(n0 + rowL0) * K + (kb) + cL0);                   \
    async_copy16((unsigned)(size_t)&Bs[buf][rowL1][cL1],                       \
                 W + (size_t)(n0 + rowL1) * K + (kb) + cL1);                   \
  } while (0)

  const int nstages = K / BK;
  ISSUE_TILE(0, 0);
  int buf = 0;

  for (int s = 0; s < nstages; ++s) {
    wait_async0();        // only stage s's 4 copies are outstanding here
    __syncthreads();      // all waves' copies for stage s have landed;
                          // also fences stage s-1's readers of buf^1
    if (s + 1 < nstages) ISSUE_TILE(buf ^ 1, (s + 1) * BK);

    // A fragments: lane<16 -> K 0..15; lane>=16 -> K 16..31
    v16h afrag[2];
    #pragma unroll
    for (int mt = 0; mt < 2; ++mt) {
      const unsigned int* rp =
          (const unsigned int*)&As[buf][wm * 32 + mt * 16 + r16][0];
      union { v8u u; v16h h; } cv;
      #pragma unroll
      for (int j = 0; j < 8; ++j) cv.u[j] = rp[hi * 8 + j];
      afrag[mt] = cv.h;
    }

    // B fragments: col = lane&15; lane<16 -> K 0..15, lane>=16 -> K 16..31
    #pragma unroll
    for (int nt = 0; nt < 4; ++nt) {
      const unsigned int* rp =
          (const unsigned int*)&Bs[buf][wn * 64 + nt * 16 + r16][0];
      union { v8u u; v16h h; } cv;
      #pragma unroll
      for (int j = 0; j < 8; ++j) cv.u[j] = rp[hi * 8 + j];
      v16h bfrag = cv.h;
      #pragma unroll
      for (int mt = 0; mt < 2; ++mt) {
        acc[mt][nt] = __builtin_amdgcn_wmma_f32_16x16x32_f16(
            false, afrag[mt], false, bfrag, (short)0, acc[mt][nt],
            false, false);
      }
    }
    buf ^= 1;
  }
#undef ISSUE_TILE

  // epilogue: C/D layout -> row = base + r + (lane>>4)*8, col = base + (lane&15)
  #pragma unroll
  for (int mt = 0; mt < 2; ++mt) {
    #pragma unroll
    for (int nt = 0; nt < 4; ++nt) {
      int col = n0 + wn * 64 + nt * 16 + r16;
      float b = bias[col];
      #pragma unroll
      for (int r = 0; r < 8; ++r) {
        int row = m0 + wm * 32 + mt * 16 + hi * 8 + r;
        C[(size_t)row * Nn + col] = acc[mt][nt][r] + b;
      }
    }
  }
}

// ---------------------------------------------------------------------------
// Sparse gathered attention: one block per (head, token), 128 threads.
// q/k/v are f32 in (N, E) row-major; col index for head h is h*DH + d.
// Duplicate samples masked (keep one survivor per distinct index), softmax
// over 128 scores, weighted gather of V, output written as f16 (N, E).
// ---------------------------------------------------------------------------
__global__ __launch_bounds__(128) void sparse_attn(
    const float* __restrict__ q, const float* __restrict__ k,
    const float* __restrict__ v, const int* __restrict__ samples,
    _Float16* __restrict__ o16)
{
  const int h   = blockIdx.x;
  const int i   = blockIdx.y;
  const int tid = threadIdx.x;            // 0..127 (sample id)

  __shared__ int   cols[SS];
  __shared__ float wgt[SS];
  __shared__ float red[SS];
  __shared__ float qs[DH];

  cols[tid] = samples[(size_t)i * SS + tid];
  if (tid < DH) qs[tid] = q[(size_t)i * EMB + h * DH + tid];
  __syncthreads();

  // duplicate detection: first occurrence survives
  const int c = cols[tid];
  bool dup = false;
  for (int t = 0; t < tid; ++t) dup |= (cols[t] == c);

  // score = q . k[col] / sqrt(N)
  const float* krow = k + (size_t)c * EMB + h * DH;
  float s = 0.f;
  #pragma unroll 8
  for (int d = 0; d < DH; ++d) s += qs[d] * krow[d];
  s = dup ? -1e30f : s * 0.03125f;        // 1/sqrt(1024)

  // block max-reduce
  red[tid] = s;
  __syncthreads();
  for (int off = 64; off > 0; off >>= 1) {
    if (tid < off) red[tid] = fmaxf(red[tid], red[tid + off]);
    __syncthreads();
  }
  float mx = red[0];
  __syncthreads();

  // exp + sum-reduce
  float e = __expf(s - mx);
  red[tid] = e;
  __syncthreads();
  for (int off = 64; off > 0; off >>= 1) {
    if (tid < off) red[tid] += red[tid + off];
    __syncthreads();
  }
  float inv = 1.f / red[0];
  __syncthreads();
  wgt[tid] = e * inv;
  __syncthreads();

  // weighted gather of V: threads 0..63 each own one d -> coalesced rows
  if (tid < DH) {
    float acc = 0.f;
    for (int s2 = 0; s2 < SS; ++s2)
      acc += wgt[s2] * v[(size_t)cols[s2] * EMB + h * DH + tid];
    o16[(size_t)i * EMB + h * DH + tid] = (_Float16)acc;
  }
}

// ---------------------------------------------------------------------------
// launcher
// ---------------------------------------------------------------------------
extern "C" void kernel_launch(void* const* d_in, const int* in_sizes, int n_in,
                              void* d_out, int out_size, void* d_ws, size_t ws_size,
                              hipStream_t stream)
{
  const float* query = (const float*)d_in[0];
  const float* key_  = (const float*)d_in[1];
  const float* value = (const float*)d_in[2];
  const float* Wq    = (const float*)d_in[3];
  const float* bq    = (const float*)d_in[4];
  const float* Wk    = (const float*)d_in[5];
  const float* bk    = (const float*)d_in[6];
  const float* Wv    = (const float*)d_in[7];
  const float* bv    = (const float*)d_in[8];
  const float* Wo    = (const float*)d_in[9];
  const float* bo    = (const float*)d_in[10];
  const int*   samples = (const int*)d_in[11];

  char* ws = (char*)d_ws;
  const size_t NE = (size_t)NTOK * EMB;          // 1M elements
  float*    q32  = (float*)(ws);                 // 4 MB
  float*    k32  = (float*)(ws + NE * 4);        // 4 MB
  float*    v32  = (float*)(ws + NE * 8);        // 4 MB
  _Float16* o16  = (_Float16*)(ws + NE * 12);    // 2 MB
  _Float16* x16  = (_Float16*)(ws + NE * 14);    // query f16
  _Float16* y16  = (_Float16*)(ws + NE * 16);    // key f16
  _Float16* z16  = (_Float16*)(ws + NE * 18);    // value f16
  _Float16* wq16 = (_Float16*)(ws + NE * 20);
  _Float16* wk16 = (_Float16*)(ws + NE * 22);
  _Float16* wv16 = (_Float16*)(ws + NE * 24);
  _Float16* wo16 = (_Float16*)(ws + NE * 26);    // total 28 MB

  const int nel = (int)NE;
  const int cgrid = (nel + 255) / 256;
  cvt_f32_to_f16<<<cgrid, 256, 0, stream>>>(query, x16, nel);
  cvt_f32_to_f16<<<cgrid, 256, 0, stream>>>(key_,  y16, nel);
  cvt_f32_to_f16<<<cgrid, 256, 0, stream>>>(value, z16, nel);
  cvt_f32_to_f16<<<cgrid, 256, 0, stream>>>(Wq, wq16, nel);
  cvt_f32_to_f16<<<cgrid, 256, 0, stream>>>(Wk, wk16, nel);
  cvt_f32_to_f16<<<cgrid, 256, 0, stream>>>(Wv, wv16, nel);
  cvt_f32_to_f16<<<cgrid, 256, 0, stream>>>(Wo, wo16, nel);

  dim3 ggrid(EMB / BN, NTOK / BM);               // 8 x 8
  gemm_wmma_f16<<<ggrid, 256, 0, stream>>>(x16, wq16, bq, q32, NTOK, EMB, EMB);
  gemm_wmma_f16<<<ggrid, 256, 0, stream>>>(y16, wk16, bk, k32, NTOK, EMB, EMB);
  gemm_wmma_f16<<<ggrid, 256, 0, stream>>>(z16, wv16, bv, v32, NTOK, EMB, EMB);

  sparse_attn<<<dim3(HEADS, NTOK), 128, 0, stream>>>(q32, k32, v32, samples, o16);

  gemm_wmma_f16<<<ggrid, 256, 0, stream>>>(o16, wo16, bo, (float*)d_out,
                                           NTOK, EMB, EMB);
}
